// AttentionEncoder_67276367725249
// MI455X (gfx1250) — matrix operand order repaired and checked by reading, not compiled, not run on hardware
//
#include <hip/hip_runtime.h>
#include <hip/hip_bf16.h>
#include <math.h>

#define B_   64
#define S_   2048
#define TRG  512
#define SRC2 1024
#define MTILE 128
#define NTILES (TRG / 16)
#define LDS_A_ELEMS (MTILE * SRC2)
#define LDS_B_ELEMS (16 * SRC2)

typedef __attribute__((ext_vector_type(16))) __bf16 v16bf;
typedef __attribute__((ext_vector_type(8)))  __bf16 v8bf;
typedef __attribute__((ext_vector_type(8)))  float  v8f;
typedef int v4i __attribute__((vector_size(16)));   // b128 payload type for async-LDS builtin

#define AS1 __attribute__((address_space(1)))
#define AS3 __attribute__((address_space(3)))

union AF { v16bf v; v8bf h[2]; };

// gfx1250 hardware tanh (V_TANH_F32) if the toolchain exposes it
#if defined(__HIP_DEVICE_COMPILE__) && __has_builtin(__builtin_amdgcn_tanhf)
#define TANH(x) __builtin_amdgcn_tanhf(x)
#else
#define TANH(x) tanhf(x)
#endif

// gfx1250 async global->LDS copy (ASYNCcnt-tracked) if exposed
#if defined(__HIP_DEVICE_COMPILE__) && \
    __has_builtin(__builtin_amdgcn_global_load_async_to_lds_b128) && \
    __has_builtin(__builtin_amdgcn_s_wait_asynccnt)
#define USE_ASYNC 1
#endif

// ---- Phase 0: W_en fp32 -> bf16 (stays hot in 192MB L2) ----
__global__ void cvt_wen_kernel(const float* __restrict__ W, __bf16* __restrict__ Wb, int n) {
  int i = blockIdx.x * 256 + threadIdx.x;
  if (i < n) Wb[i] = (__bf16)W[i];
}

// ---- Phase 1: de_proj[b,h] = dehy[b,:] . W_de[h,:] + b_en[h] ----
__global__ void __launch_bounds__(TRG)
deproj_kernel(const float* __restrict__ dehy, const float* __restrict__ W_de,
              const float* __restrict__ b_en, float* __restrict__ de_proj) {
  const int b = blockIdx.x, h = threadIdx.x;
  const float* x = dehy + b * TRG;
  const float* w = W_de + (size_t)h * TRG;
  float acc = b_en[h];
#pragma unroll 4
  for (int k = 0; k < TRG; ++k) acc += x[k] * w[k];
  de_proj[b * TRG + h] = acc;
}

// Stage one 16x1024 bf16 B-tile (32 KB): thread t copies 8 x 16B.
__device__ __forceinline__ void stage_b_tile(const __bf16* __restrict__ src_base,
                                             __bf16* dst, int t) {
  const int row = t >> 4;            // 0..15
  const int k0  = (t & 15) * 64;
  const __bf16* src = src_base + (size_t)row * SRC2 + k0;
  __bf16* d = dst + row * SRC2 + k0;
#ifdef USE_ASYNC
#pragma unroll
  for (int k = 0; k < 64; k += 8)
    __builtin_amdgcn_global_load_async_to_lds_b128(
        (AS1 v4i*)(void*)(src + k), (AS3 v4i*)(void*)(d + k), 0, 0);
#else
#pragma unroll
  for (int k = 0; k < 64; k += 8)
    *(v8bf*)(d + k) = *(const v8bf*)(src + k);
#endif
}

// ---- Phase 2: fused score GEMM (bf16 WMMA, fp32 acc) + tanh/coverage epilogue ----
__global__ void __launch_bounds__(256)
score_kernel(const float* __restrict__ enhy, const float* __restrict__ past_attn,
             const float* __restrict__ de_proj, const float* __restrict__ w_cv,
             const float* __restrict__ w_warp, const __bf16* __restrict__ Wb,
             float* __restrict__ scores) {
  extern __shared__ __bf16 lds[];
  __bf16* As = lds;                                    // [128][1024] = 256 KB
  __bf16* Bbuf0 = lds + LDS_A_ELEMS;                   // [16][1024]  =  32 KB
  __bf16* Bbuf1 = lds + LDS_A_ELEMS + LDS_B_ELEMS;     // [16][1024]  =  32 KB

  const int b    = blockIdx.y;
  const int s0   = blockIdx.x * MTILE;
  const int t    = threadIdx.x;
  const int wave = t >> 5, lane = t & 31;
  const int half = lane >> 4, l16 = lane & 15;

  // Stage A: 2 threads per row, 512 cols each; convert fp32 -> bf16.
  {
    const int row = t >> 1;
    const int k0  = (t & 1) * 512;
    const float* src = enhy + ((size_t)(b * S_ + s0 + row)) * SRC2 + k0;
    __bf16* dst = As + row * SRC2 + k0;
#pragma unroll 8
    for (int k = 0; k < 512; k += 8) {
      float4 f0 = *(const float4*)(src + k);
      float4 f1 = *(const float4*)(src + k + 4);
      v8bf o = { (__bf16)f0.x, (__bf16)f0.y, (__bf16)f0.z, (__bf16)f0.w,
                 (__bf16)f1.x, (__bf16)f1.y, (__bf16)f1.z, (__bf16)f1.w };
      *(v8bf*)(dst + k) = o;
    }
  }

  // past_attn for this lane's 8 output rows (C-tile layout: M = j + 8*half)
  const int rbase = s0 + wave * 16 + half * 8;
  float pa[8];
#pragma unroll
  for (int j = 0; j < 8; ++j) pa[j] = past_attn[b * S_ + rbase + j];

  float rowacc[8] = {0.f, 0.f, 0.f, 0.f, 0.f, 0.f, 0.f, 0.f};
  const __bf16* aptr = As + (wave * 16 + l16) * SRC2 + half * 8;

  // Preload B tile 0 (async on gfx1250: GLOBAL_LOAD_ASYNC_TO_LDS_B128)
  stage_b_tile(Wb, Bbuf0, t);

  for (int nc = 0; nc < NTILES; ++nc) {
    const int cur = nc & 1;
    __bf16* Bcur = cur ? Bbuf1 : Bbuf0;
    __bf16* Bnxt = cur ? Bbuf0 : Bbuf1;
#ifdef USE_ASYNC
    __builtin_amdgcn_s_wait_asynccnt(0);   // my tile-nc async loads landed
#endif
    __syncthreads();                       // all waves: loads landed, prev compute done
    if (nc + 1 < NTILES)                   // overlap next tile load with compute
      stage_b_tile(Wb + (size_t)(nc + 1) * 16 * SRC2, Bnxt, t);

    v8f c = {};
    const __bf16* bptr = Bcur + l16 * SRC2 + half * 8;
#pragma unroll 4
    for (int kk = 0; kk < SRC2; kk += 32) {
      AF a, bb;
      a.h[0]  = *(const v8bf*)(aptr + kk);        // A: K = kk + 8*half .. +7
      a.h[1]  = *(const v8bf*)(aptr + kk + 16);   // A: K = kk+16+8*half .. +7
      bb.h[0] = *(const v8bf*)(bptr + kk);        // B: K = kk + 8*half .. +7
      bb.h[1] = *(const v8bf*)(bptr + kk + 16);
      c = __builtin_amdgcn_wmma_f32_16x16x32_bf16(false, a.v, false, bb.v,
                                                  (short)0, c, false, false);
    }

    // epilogue: tanh(c + de_proj + past_attn*w_cv) * w_warp, partial N-reduction
    const int n = nc * 16 + l16;
    const float dp = de_proj[b * TRG + n];
    const float wc = w_cv[n];
    const float ww = w_warp[n];
#pragma unroll
    for (int j = 0; j < 8; ++j)
      rowacc[j] += ww * TANH(c[j] + dp + pa[j] * wc);
  }

  // reduce over the 16 N-lanes of each wave32 half
#pragma unroll
  for (int j = 0; j < 8; ++j) {
    float v = rowacc[j];
    v += __shfl_xor(v, 1);
    v += __shfl_xor(v, 2);
    v += __shfl_xor(v, 4);
    v += __shfl_xor(v, 8);
    rowacc[j] = v;
  }
  if (l16 == 0) {                      // lanes 0 (rows 0..7) and 16 (rows 8..15)
#pragma unroll
    for (int j = 0; j < 8; ++j)
      scores[b * S_ + rbase + j] = rowacc[j];
  }
}

// ---- Phase 3: mask + softmax over S (one block per batch) ----
__global__ void __launch_bounds__(256)
softmax_kernel(const float* __restrict__ scores, const int* __restrict__ mask,
               float* __restrict__ attn, float* __restrict__ attn_ee) {
  const int b = blockIdx.x, t = threadIdx.x;
  __shared__ float red[8];
  float vals[8];
  float lmax = -INFINITY;
#pragma unroll
  for (int i = 0; i < 8; ++i) {
    const int idx = b * S_ + t + i * 256;
    float v = scores[idx];
    if (mask[idx] == 0) v = -1e20f;
    vals[i] = v;
    attn_ee[idx] = v;
    lmax = fmaxf(lmax, v);
  }
  for (int m = 16; m >= 1; m >>= 1) lmax = fmaxf(lmax, __shfl_xor(lmax, m));
  if ((t & 31) == 0) red[t >> 5] = lmax;
  __syncthreads();
  float gmax = red[0];
#pragma unroll
  for (int i = 1; i < 8; ++i) gmax = fmaxf(gmax, red[i]);
  __syncthreads();
  float e[8];
  float lsum = 0.f;
#pragma unroll
  for (int i = 0; i < 8; ++i) { e[i] = __expf(vals[i] - gmax); lsum += e[i]; }
  for (int m = 16; m >= 1; m >>= 1) lsum += __shfl_xor(lsum, m);
  if ((t & 31) == 0) red[t >> 5] = lsum;
  __syncthreads();
  float gsum = 0.f;
#pragma unroll
  for (int i = 0; i < 8; ++i) gsum += red[i];
  const float inv = 1.f / gsum;
#pragma unroll
  for (int i = 0; i < 8; ++i) attn[b * S_ + t + i * 256] = e[i] * inv;
}

// ---- Phase 4: c_encoder[b,d] = sum_s attn[b,s] * enhy[b,s,d] ----
__global__ void __launch_bounds__(1024)
context_kernel(const float* __restrict__ attn, const float* __restrict__ enhy,
               float* __restrict__ cenc) {
  const int b = blockIdx.x, d = threadIdx.x;
  __shared__ float a_sh[S_];
  for (int i = threadIdx.x; i < S_; i += 1024) a_sh[i] = attn[b * S_ + i];
  __syncthreads();
  float acc = 0.f;
  const float* base = enhy + (size_t)b * S_ * SRC2 + d;
#pragma unroll 4
  for (int s = 0; s < S_; ++s) acc += a_sh[s] * base[(size_t)s * SRC2];
  cenc[b * SRC2 + d] = acc;
}

extern "C" void kernel_launch(void* const* d_in, const int* in_sizes, int n_in,
                              void* d_out, int out_size, void* d_ws, size_t ws_size,
                              hipStream_t stream) {
  const float* dehy      = (const float*)d_in[0];
  const float* enhy      = (const float*)d_in[1];
  const float* past_attn = (const float*)d_in[2];
  const int*   src_mask  = (const int*)  d_in[3];
  const float* W_en      = (const float*)d_in[4];
  const float* b_en      = (const float*)d_in[5];
  const float* W_de      = (const float*)d_in[6];
  const float* w_cv      = (const float*)d_in[7];
  const float* w_warp    = (const float*)d_in[8];

  float* out_cenc = (float*)d_out;           // [64,1024]
  float* out_attn = out_cenc + B_ * SRC2;    // [64,2048]
  float* out_ee   = out_attn + B_ * S_;      // [64,2048]

  char*   ws      = (char*)d_ws;
  __bf16* Wb      = (__bf16*)ws;                               // 1 MB
  float*  de_proj = (float*)(ws + (size_t)TRG * SRC2 * 2);     // 128 KB
  float*  scores  = de_proj + B_ * TRG;                        // 512 KB

  cvt_wen_kernel<<<(TRG * SRC2 + 255) / 256, 256, 0, stream>>>(W_en, Wb, TRG * SRC2);
  deproj_kernel<<<B_, TRG, 0, stream>>>(dehy, W_de, b_en, de_proj);

  dim3 grid(S_ / MTILE, B_);
  const size_t ldsBytes = (size_t)(LDS_A_ELEMS + 2 * LDS_B_ELEMS) * sizeof(__bf16); // 320 KB
  score_kernel<<<grid, 256, ldsBytes, stream>>>(enhy, past_attn, de_proj, w_cv, w_warp,
                                                Wb, scores);
  softmax_kernel<<<B_, 256, 0, stream>>>(scores, src_mask, out_attn, out_ee);
  context_kernel<<<B_, 1024, 0, stream>>>(out_attn, enhy, out_cenc);
}
